// InnerproductSimilarity_80444737454122
// MI455X (gfx1250) — matrix operand order repaired and checked by reading, not compiled, not run on hardware
//
#include <hip/hip_runtime.h>
#include <hip/hip_bf16.h>

typedef __attribute__((ext_vector_type(16))) _Float16 v16h;
typedef __attribute__((ext_vector_type(8)))  _Float16 v8h;
typedef __attribute__((ext_vector_type(8)))  float    v8f;

// Problem constants
#define NQ     32      // queries
#define NWAY   5       // classes
#define C      64      // channels (K of the GEMM)
#define PQ     441     // query positions (21*21)
#define PS     2205    // support cols per class (5*441)
#define PQ_PAD 448     // 28 * 16
#define PS_PAD 2208    // 138 * 16
#define RT     28      // row tiles per query
#define CT     138     // col tiles per class
#define TILES_PER_PAIR (RT * CT)          // 3864
#define NPAIR  (NQ * NWAY)                // 160
#define NTILES (NPAIR * TILES_PER_PAIR)   // 618240

// ---------------------------------------------------------------------------
// Stage 1a: L2-normalize query over channels, emit f16 A matrix
// A[(q*448 + i) * 64 + c], rows i in [441,448) zero-padded.
// query_xf flat: [32, 64, 441]
// ---------------------------------------------------------------------------
__global__ void prep_query(const float* __restrict__ qin, _Float16* __restrict__ A) {
    int tid = blockIdx.x * blockDim.x + threadIdx.x;
    if (tid >= NQ * PQ_PAD) return;
    int q = tid / PQ_PAD;
    int i = tid - q * PQ_PAD;
    bool valid = (i < PQ);

    float v[C];
    float ss = 0.0f;
#pragma unroll
    for (int c = 0; c < C; ++c) {
        float x = valid ? qin[((size_t)q * C + c) * PQ + i] : 0.0f;
        v[c] = x;
        ss += x * x;
    }
    float s = 1.0f / fmaxf(sqrtf(ss), 1e-12f);

    _Float16* dst = A + (size_t)tid * C;
#pragma unroll
    for (int ch = 0; ch < C / 8; ++ch) {
        v8h h;
#pragma unroll
        for (int e = 0; e < 8; ++e) h[e] = (_Float16)(v[ch * 8 + e] * s);
        *(v8h*)(dst + ch * 8) = h;
    }
}

// ---------------------------------------------------------------------------
// Stage 1b: L2-normalize support over channels, emit f16 B^T matrix
// Bt[(n*2208 + j) * 64 + c], cols j in [2205,2208) zero-padded.
// support_xf flat: [25, 64, 441]; j = k*441 + p maps shot k, position p.
// ---------------------------------------------------------------------------
__global__ void prep_support(const float* __restrict__ sin_, _Float16* __restrict__ Bt) {
    int tid = blockIdx.x * blockDim.x + threadIdx.x;
    if (tid >= NWAY * PS_PAD) return;
    int n = tid / PS_PAD;
    int j = tid - n * PS_PAD;
    bool valid = (j < PS);
    int k = j / PQ;          // shot
    int p = j - k * PQ;      // spatial position

    float v[C];
    float ss = 0.0f;
#pragma unroll
    for (int c = 0; c < C; ++c) {
        float x = valid ? sin_[(((size_t)(n * 5 + k)) * C + c) * PQ + p] : 0.0f;
        v[c] = x;
        ss += x * x;
    }
    float s = 1.0f / fmaxf(sqrtf(ss), 1e-12f);

    _Float16* dst = Bt + (size_t)tid * C;
#pragma unroll
    for (int ch = 0; ch < C / 8; ++ch) {
        v8h h;
#pragma unroll
        for (int e = 0; e < 8; ++e) h[e] = (_Float16)(v[ch * 8 + e] * s);
        *(v8h*)(dst + ch * 8) = h;
    }
}

// ---------------------------------------------------------------------------
// Stage 2: tiled GEMM with v_wmma_f32_16x16x32_f16.
// One wave = one 16x16 output tile, K=64 = two WMMAs.
// Fragment gathers follow the CDNA5 ISA VGPR layouts:
//   A 16x32 f16 : lane<16 -> M=lane,    halves 0-7 = K(kb..kb+7),   8-15 = K(kb+16..kb+23), kb = kt*32
//                 lane>=16 -> M=lane-16, halves 0-7 = K(kb+8..15),   8-15 = K(kb+24..31)
//   B 32x16 f16 : lane<16 -> N=lane,    halves 0-7 = K(kb..kb+7),   8-15 = K(kb+16..kb+23)
//                 lane>=16 -> N=lane-16, halves 0-7 = K(kb+8..15),   8-15 = K(kb+24..31)
//   C/D 16x16 f32: VGPR r, lane<16 -> (M=r,   N=lane); lane>=16 -> (M=r+8, N=lane-16)
// Epilogue is specialized: interior tiles (~96%) store unconditionally via a
// single base pointer + immediate offsets; only rt==27 / ct==137 tiles guard.
// ---------------------------------------------------------------------------
__global__ void simi_wmma(const _Float16* __restrict__ A,
                          const _Float16* __restrict__ Bt,
                          float* __restrict__ out) {
    const int lane   = threadIdx.x & 31;
    const int laneLo = lane & 15;
    const int hi     = lane >> 4;                       // 0 or 1
    const int tile   = blockIdx.x * (blockDim.x >> 5) + (threadIdx.x >> 5);

    int pair = tile / TILES_PER_PAIR;
    int rem  = tile - pair * TILES_PER_PAIR;
    int rt   = rem / CT;
    int ct   = rem - rt * CT;
    int q    = pair / NWAY;
    int n    = pair - q * NWAY;

    const _Float16* arow = A  + (size_t)(q * PQ_PAD + rt * 16 + laneLo) * C;
    const _Float16* brow = Bt + (size_t)(n * PS_PAD + ct * 16 + laneLo) * C;

    v8f acc = {};
#pragma unroll
    for (int kt = 0; kt < 2; ++kt) {
        const int kbA = kt * 32 + hi * 8;
        v8h a_lo = *(const v8h*)(arow + kbA);
        v8h a_hi = *(const v8h*)(arow + kbA + 16);
        v16h a = __builtin_shufflevector(a_lo, a_hi,
                 0,1,2,3,4,5,6,7,8,9,10,11,12,13,14,15);

        const int kbB = kt * 32 + hi * 8;
        v8h b_lo = *(const v8h*)(brow + kbB);
        v8h b_hi = *(const v8h*)(brow + kbB + 16);
        v16h b = __builtin_shufflevector(b_lo, b_hi,
                 0,1,2,3,4,5,6,7,8,9,10,11,12,13,14,15);

        acc = __builtin_amdgcn_wmma_f32_16x16x32_f16(
                  /*neg_a=*/false, a, /*neg_b=*/false, b,
                  /*c_mod=*/(short)0, acc, /*reuse_a=*/false, /*reuse_b=*/false);
    }

    // Write (1 + sim) * 0.5 into out[q][n][i][j]
    const int j  = ct * 16 + laneLo;
    const int i0 = rt * 16 + hi * 8;
    float* base = out + ((size_t)(q * NWAY + n) * PQ) * PS;

    if (rt < RT - 1 && ct < CT - 1) {
        // Interior tile: all 16 rows < 441, all 16 cols < 2205.
        // One address, 8 stores at immediate offsets (8820*r bytes < 2^23).
        float* p = base + (size_t)i0 * PS + j;
#pragma unroll
        for (int r = 0; r < 8; ++r)
            p[(size_t)r * PS] = (1.0f + acc[r]) * 0.5f;
    } else {
        // Boundary tile: guard rows/cols.
        if (j < PS) {
#pragma unroll
            for (int r = 0; r < 8; ++r) {
                int i = i0 + r;
                if (i < PQ)
                    base[(size_t)i * PS + j] = (1.0f + acc[r]) * 0.5f;
            }
        }
    }
}

// ---------------------------------------------------------------------------
extern "C" void kernel_launch(void* const* d_in, const int* in_sizes, int n_in,
                              void* d_out, int out_size, void* d_ws, size_t ws_size,
                              hipStream_t stream) {
    // Inputs (setup_inputs order): support_xf f32, support_y i64 (unused),
    //                              query_xf f32,   query_y i64 (unused)
    const float* support_xf = (const float*)d_in[0];
    const float* query_xf   = (const float*)d_in[2];
    float* out = (float*)d_out;

    // Workspace carve-out (both 256B aligned)
    _Float16* A  = (_Float16*)d_ws;                                  // 32*448*64*2 = 1,835,008 B
    _Float16* Bt = (_Float16*)((char*)d_ws + (size_t)NQ * PQ_PAD * C * sizeof(_Float16));
                                                                     // 5*2208*64*2 = 1,413,120 B

    // Stage 1: normalize + f16 pack (padded)
    {
        int nth = NQ * PQ_PAD;                 // 14336
        prep_query<<<(nth + 255) / 256, 256, 0, stream>>>(query_xf, A);
    }
    {
        int nth = NWAY * PS_PAD;               // 11040
        prep_support<<<(nth + 255) / 256, 256, 0, stream>>>(support_xf, Bt);
    }

    // Stage 2: WMMA GEMM, 8 waves (tiles) per 256-thread block
    {
        int blocks = NTILES / 8;               // 618240 / 8 = 77280, exact
        simi_wmma<<<blocks, 256, 0, stream>>>(A, Bt, out);
    }
    (void)in_sizes; (void)n_in; (void)out_size; (void)ws_size;
}
